// MMFP4FusedMoE_30915174596914
// MI455X (gfx1250) — matrix operand order repaired and checked by reading, not compiled, not run on hardware
//
#include <hip/hip_runtime.h>
#include <cstdint>

// Problem constants (from reference)
constexpr int TE   = 16;    // experts
constexpr int TOPK = 4;
constexpr int HD   = 2048;  // hidden
constexpr int ID   = 1536;  // intermediate
constexpr int GSZ  = 128;   // fp4 scale group size (== 4 K-steps of 32)
constexpr int NT   = 1024;  // tokens (B*S)

typedef __attribute__((ext_vector_type(16))) _Float16 v16h;
typedef __attribute__((ext_vector_type(8)))  float    v8f;

union AV { uint4    q[2]; v16h v; };
union BV { uint32_t u[8]; v16h v; };

// ---------------------------------------------------------------------------
// FP4(E2M1) -> F16 dequant of one packed word (8 nibbles, nibble0 = lowest).
// F16 of fp4 magnitudes has a zero low byte, so everything is byte shuffles:
//   mags 0..7 -> hi bytes {00,38,3C,3E,40,42,44,46}; sign = nibble bit3 -> 0x80.
// out[0..3] = {f16(K0),f16(K1)} .. {f16(K6),f16(K7)} packed 2 per dword.
// ---------------------------------------------------------------------------
__device__ __forceinline__ void dq_word(uint32_t w, uint32_t out[4]) {
  uint32_t en = w & 0x0F0F0F0Fu;          // nibbles 0,2,4,6 -> K0,K2,K4,K6
  uint32_t on = (w >> 4) & 0x0F0F0F0Fu;   // nibbles 1,3,5,7 -> K1,K3,K5,K7
  uint32_t eh = __builtin_amdgcn_perm(0x46444240u, 0x3E3C3800u, en & 0x07070707u)
              | ((en & 0x08080808u) << 4);
  uint32_t oh = __builtin_amdgcn_perm(0x46444240u, 0x3E3C3800u, on & 0x07070707u)
              | ((on & 0x08080808u) << 4);
  uint32_t e02 = __builtin_amdgcn_perm(0u, eh, 0x01040004u); // {f16K0,f16K2}
  uint32_t e46 = __builtin_amdgcn_perm(0u, eh, 0x03040204u); // {f16K4,f16K6}
  uint32_t o13 = __builtin_amdgcn_perm(0u, oh, 0x01040004u);
  uint32_t o57 = __builtin_amdgcn_perm(0u, oh, 0x03040204u);
  out[0] = __builtin_amdgcn_perm(o13, e02, 0x05040100u);     // {K0,K1}
  out[1] = __builtin_amdgcn_perm(o13, e02, 0x07060302u);     // {K2,K3}
  out[2] = __builtin_amdgcn_perm(o57, e46, 0x05040100u);     // {K4,K5}
  out[3] = __builtin_amdgcn_perm(o57, e46, 0x07060302u);     // {K6,K7}
}

// ---------------------------------------------------------------------------
// K1: routing. One thread per token: logits = x . gate_w[e], top-4 of logits,
// renormalized softmax weights over the selected 4 (softmax denom cancels).
// ---------------------------------------------------------------------------
__global__ __launch_bounds__(256) void route_kernel(
    const float* __restrict__ x, const float* __restrict__ gw,
    int* __restrict__ tok_idx, float* __restrict__ tok_w) {
  int t = blockIdx.x * 256 + threadIdx.x;
  if (t >= NT) return;
  float acc[TE];
#pragma unroll
  for (int e = 0; e < TE; ++e) acc[e] = 0.f;
  for (int h = 0; h < HD; h += 4) {
    float4 xv = *reinterpret_cast<const float4*>(x + (size_t)t * HD + h);
#pragma unroll
    for (int e = 0; e < TE; ++e) {
      float4 g = *reinterpret_cast<const float4*>(gw + (size_t)e * HD + h);
      acc[e] += xv.x * g.x + xv.y * g.y + xv.z * g.z + xv.w * g.w;
    }
  }
  int   bi[TOPK];
  float bv[TOPK];
  bool used[TE];
#pragma unroll
  for (int e = 0; e < TE; ++e) used[e] = false;
#pragma unroll
  for (int k = 0; k < TOPK; ++k) {
    float best = -3.4e38f; int b = 0;
#pragma unroll
    for (int e = 0; e < TE; ++e)
      if (!used[e] && acc[e] > best) { best = acc[e]; b = e; }
    used[b] = true; bi[k] = b; bv[k] = best;
  }
  float m = bv[0], s = 0.f, w[TOPK];
#pragma unroll
  for (int k = 0; k < TOPK; ++k) { w[k] = __expf(bv[k] - m); s += w[k]; }
  float inv = 1.f / s;
#pragma unroll
  for (int k = 0; k < TOPK; ++k) {
    tok_idx[t * TOPK + k] = bi[k];
    tok_w[t * TOPK + k]   = w[k] * inv;
  }
}

// ---------------------------------------------------------------------------
// K2: deterministic compaction into per-expert token lists.
// 1024 threads (one per token) = 32 waves; rank = wave ballot + block prefix.
// ---------------------------------------------------------------------------
__global__ __launch_bounds__(1024) void compact_kernel(
    const int* __restrict__ tok_idx, const float* __restrict__ tok_w,
    int* __restrict__ counts, int* __restrict__ offsets,
    int* __restrict__ tokens_list, float* __restrict__ weight_list) {
  const int t = threadIdx.x, lane = t & 31, wv = t >> 5;
  int idx[TOPK]; float wgt[TOPK]; int myrank[TOPK] = {0, 0, 0, 0};
#pragma unroll
  for (int k = 0; k < TOPK; ++k) {
    idx[k] = tok_idx[t * TOPK + k];
    wgt[k] = tok_w[t * TOPK + k];
  }
  __shared__ int wcnt[TE][32];
  __shared__ int cnt_s[TE];
  __shared__ int off_s[TE];
  for (int e = 0; e < TE; ++e) {
    bool pred = (idx[0] == e) | (idx[1] == e) | (idx[2] == e) | (idx[3] == e);
    uint32_t mask = __builtin_amdgcn_ballot_w32(pred);
    int r = __popc(mask & ((1u << lane) - 1u));
#pragma unroll
    for (int k = 0; k < TOPK; ++k)
      if (idx[k] == e) myrank[k] = r;
    if (lane == 0) wcnt[e][wv] = __popc(mask);
  }
  __syncthreads();
  if (t < TE) {  // exclusive scan of per-wave counts for expert t
    int s = 0;
    for (int w2 = 0; w2 < 32; ++w2) { int c = wcnt[t][w2]; wcnt[t][w2] = s; s += c; }
    cnt_s[t] = s; counts[t] = s;
  }
  __syncthreads();
  if (t == 0) {
    int s = 0;
    for (int e = 0; e < TE; ++e) { off_s[e] = s; offsets[e] = s; s += cnt_s[e]; }
  }
  __syncthreads();
#pragma unroll
  for (int k = 0; k < TOPK; ++k) {
    int e = idx[k];
    int pos = off_s[e] + wcnt[e][wv] + myrank[k];
    tokens_list[pos] = t;
    weight_list[pos] = wgt[k];
  }
}

// ---------------------------------------------------------------------------
// K3: grouped gate_up GEMM + SwiGLU. Block: 32 tokens x 128 I-columns,
// 8 waves each own two 16x16 M-tiles sharing every dequanted B fragment
// (halves per-FLOP dequant cost). Double-buffered LDS A staging: one
// __syncthreads per K-step. Per-group (128 H) fp32 scales folded into f32
// accumulators so the F16 B-tiles stay numerically exact.
// ---------------------------------------------------------------------------
template <bool SHARED>
__global__ __launch_bounds__(256) void gateup_kernel(
    const float* __restrict__ x,
    const int* __restrict__ packed, const float* __restrict__ scales,
    const int* __restrict__ counts, const int* __restrict__ offsets,
    const int* __restrict__ tokens_list, float* __restrict__ act) {
  const int e     = SHARED ? 0 : blockIdx.z;
  const int cnt   = SHARED ? NT : counts[e];
  const int base  = SHARED ? 0 : offsets[e];
  const int mtile = blockIdx.y;
  if (mtile * 32 >= cnt) return;  // uniform exit (cnt is block-uniform)

  const int lane = threadIdx.x & 31;
  const int wv   = threadIdx.x >> 5;
  const int twoI = 2 * ID;
  const int n0   = blockIdx.x * 128 + wv * 16;
  const int colg = n0 + (lane & 15);
  const int colu = colg + ID;
  const int* pk   = packed + (SHARED ? 0 : (size_t)e * (HD / 8) * twoI);
  const float* sc = scales + (SHARED ? 0 : (size_t)e * (HD / GSZ) * twoI);

  __shared__ __align__(16) _Float16 aS[2][32 * 40];  // 40-half row stride: 16B aligned

  v8f acc_g0 = {}, acc_g1 = {}, acc_u0 = {}, acc_u1 = {};
  v8f blk_g0 = {}, blk_g1 = {}, blk_u0 = {}, blk_u1 = {};
  const int m16 = lane & 15;
  const int kA  = (lane >> 4) * 16;  // A-fragment K base (contiguous: k = 16h + s)
  const int wro = (lane >> 4) * 2;   // B: packed-word row offset ((lane>>4)*16)/8

  auto stageA = [&](int h0, _Float16* buf) {
#pragma unroll
    for (int j = 0; j < 4; ++j) {
      int q = threadIdx.x + 256 * j;     // 0..1023
      int m = q >> 5, k = q & 31;
      int row = mtile * 32 + m;
      float v = 0.f;
      if (row < cnt) {
        int tok = SHARED ? row : tokens_list[base + row];
        v = x[(size_t)tok * HD + h0 + k];
      }
      buf[m * 40 + k] = (_Float16)v;
    }
  };

  stageA(0, aS[0]);
  int p = 0;
  for (int h0 = 0; h0 < HD; h0 += 32, p ^= 1) {
    __syncthreads();
    if (h0 + 32 < HD) stageA(h0 + 32, aS[p ^ 1]);

    const _Float16* buf = aS[p];
    AV a0, a1;
    a0.q[0] = *reinterpret_cast<const uint4*>(&buf[m16 * 40 + kA]);
    a0.q[1] = *reinterpret_cast<const uint4*>(&buf[m16 * 40 + kA + 8]);
    a1.q[0] = *reinterpret_cast<const uint4*>(&buf[(m16 + 16) * 40 + kA]);
    a1.q[1] = *reinterpret_cast<const uint4*>(&buf[(m16 + 16) * 40 + kA + 8]);

    int wrow = (h0 >> 3) + wro;
    uint32_t g0 = (uint32_t)pk[(size_t)wrow * twoI + colg];
    uint32_t g1 = (uint32_t)pk[(size_t)(wrow + 1) * twoI + colg];
    uint32_t u0 = (uint32_t)pk[(size_t)wrow * twoI + colu];
    uint32_t u1 = (uint32_t)pk[(size_t)(wrow + 1) * twoI + colu];
    BV bg, bu;
    dq_word(g0, bg.u); dq_word(g1, bg.u + 4);
    dq_word(u0, bu.u); dq_word(u1, bu.u + 4);

    blk_g0 = __builtin_amdgcn_wmma_f32_16x16x32_f16(false, a0.v, false, bg.v,
                                                    (short)0, blk_g0, false, false);
    blk_g1 = __builtin_amdgcn_wmma_f32_16x16x32_f16(false, a1.v, false, bg.v,
                                                    (short)0, blk_g1, false, false);
    blk_u0 = __builtin_amdgcn_wmma_f32_16x16x32_f16(false, a0.v, false, bu.v,
                                                    (short)0, blk_u0, false, false);
    blk_u1 = __builtin_amdgcn_wmma_f32_16x16x32_f16(false, a1.v, false, bu.v,
                                                    (short)0, blk_u1, false, false);

    if (((h0 >> 5) & 3) == 3) {  // end of a 128-wide H group: fold fp32 scales
      int grp = h0 >> 7;
      float sg = sc[(size_t)grp * twoI + colg];
      float su = sc[(size_t)grp * twoI + colu];
#pragma unroll
      for (int r = 0; r < 8; ++r) {
        acc_g0[r] += sg * blk_g0[r]; blk_g0[r] = 0.f;
        acc_g1[r] += sg * blk_g1[r]; blk_g1[r] = 0.f;
        acc_u0[r] += su * blk_u0[r]; blk_u0[r] = 0.f;
        acc_u1[r] += su * blk_u1[r]; blk_u1[r] = 0.f;
      }
    }
  }

  // SwiGLU and store act (f32). D layout: lanes 0-15 -> M=r, lanes 16-31 -> M=r+8
  const int mhalf = (lane >> 4) * 8;
#pragma unroll
  for (int r = 0; r < 8; ++r) {
    int row0 = mtile * 32 + mhalf + r;
    int row1 = row0 + 16;
    if (row0 < cnt) {
      float g = acc_g0[r], u = acc_u0[r];
      float s = __fdividef(g, 1.f + __expf(-g));
      act[(size_t)(base + row0) * ID + colg] = s * u;
    }
    if (row1 < cnt) {
      float g = acc_g1[r], u = acc_u1[r];
      float s = __fdividef(g, 1.f + __expf(-g));
      act[(size_t)(base + row1) * ID + colg] = s * u;
    }
  }
}

// ---------------------------------------------------------------------------
// K4: down GEMM (same 32-token x 128-col tiling, B shared by two M-tiles).
// SHARED: writes out (covers every element -> no memset needed).
// Routed: atomicAdd(combine_weight * y) into out.
// ---------------------------------------------------------------------------
template <bool SHARED>
__global__ __launch_bounds__(256) void down_kernel(
    const float* __restrict__ act,
    const int* __restrict__ packed, const float* __restrict__ scales,
    const int* __restrict__ counts, const int* __restrict__ offsets,
    const int* __restrict__ tokens_list, const float* __restrict__ wts,
    float* __restrict__ out) {
  const int e     = SHARED ? 0 : blockIdx.z;
  const int cnt   = SHARED ? NT : counts[e];
  const int base  = SHARED ? 0 : offsets[e];
  const int mtile = blockIdx.y;
  if (mtile * 32 >= cnt) return;

  const int lane = threadIdx.x & 31;
  const int wv   = threadIdx.x >> 5;
  const int n0   = blockIdx.x * 128 + wv * 16;
  const int col  = n0 + (lane & 15);  // H column
  const int* pk   = packed + (SHARED ? 0 : (size_t)e * (ID / 8) * HD);
  const float* sc = scales + (SHARED ? 0 : (size_t)e * (ID / GSZ) * HD);

  __shared__ __align__(16) _Float16 aS[2][32 * 40];

  v8f acc0 = {}, acc1 = {}, blk0 = {}, blk1 = {};
  const int m16 = lane & 15;
  const int kA  = (lane >> 4) * 16;
  const int wro = (lane >> 4) * 2;

  auto stageA = [&](int i0, _Float16* buf) {
#pragma unroll
    for (int j = 0; j < 4; ++j) {
      int q = threadIdx.x + 256 * j;
      int m = q >> 5, k = q & 31;
      int row = mtile * 32 + m;
      float v = 0.f;
      if (row < cnt) v = act[(size_t)(base + row) * ID + i0 + k];
      buf[m * 40 + k] = (_Float16)v;
    }
  };

  stageA(0, aS[0]);
  int p = 0;
  for (int i0 = 0; i0 < ID; i0 += 32, p ^= 1) {
    __syncthreads();
    if (i0 + 32 < ID) stageA(i0 + 32, aS[p ^ 1]);

    const _Float16* buf = aS[p];
    AV a0, a1;
    a0.q[0] = *reinterpret_cast<const uint4*>(&buf[m16 * 40 + kA]);
    a0.q[1] = *reinterpret_cast<const uint4*>(&buf[m16 * 40 + kA + 8]);
    a1.q[0] = *reinterpret_cast<const uint4*>(&buf[(m16 + 16) * 40 + kA]);
    a1.q[1] = *reinterpret_cast<const uint4*>(&buf[(m16 + 16) * 40 + kA + 8]);

    int wrow = (i0 >> 3) + wro;
    uint32_t w0 = (uint32_t)pk[(size_t)wrow * HD + col];
    uint32_t w1 = (uint32_t)pk[(size_t)(wrow + 1) * HD + col];
    BV b;
    dq_word(w0, b.u); dq_word(w1, b.u + 4);

    blk0 = __builtin_amdgcn_wmma_f32_16x16x32_f16(false, a0.v, false, b.v,
                                                  (short)0, blk0, false, false);
    blk1 = __builtin_amdgcn_wmma_f32_16x16x32_f16(false, a1.v, false, b.v,
                                                  (short)0, blk1, false, false);

    if (((i0 >> 5) & 3) == 3) {
      int grp = i0 >> 7;
      float s = sc[(size_t)grp * HD + col];
#pragma unroll
      for (int r = 0; r < 8; ++r) {
        acc0[r] += s * blk0[r]; blk0[r] = 0.f;
        acc1[r] += s * blk1[r]; blk1[r] = 0.f;
      }
    }
  }

  const int mhalf = (lane >> 4) * 8;
#pragma unroll
  for (int r = 0; r < 8; ++r) {
    int row0 = mtile * 32 + mhalf + r;
    int row1 = row0 + 16;
    if (row0 < cnt) {
      if (SHARED) {
        out[(size_t)row0 * HD + col] = acc0[r];  // full overwrite, runs first
      } else {
        int tok = tokens_list[base + row0];
        atomicAdd(&out[(size_t)tok * HD + col], wts[base + row0] * acc0[r]);
      }
    }
    if (row1 < cnt) {
      if (SHARED) {
        out[(size_t)row1 * HD + col] = acc1[r];
      } else {
        int tok = tokens_list[base + row1];
        atomicAdd(&out[(size_t)tok * HD + col], wts[base + row1] * acc1[r]);
      }
    }
  }
}

// ---------------------------------------------------------------------------
extern "C" void kernel_launch(void* const* d_in, const int* in_sizes, int n_in,
                              void* d_out, int out_size, void* d_ws, size_t ws_size,
                              hipStream_t stream) {
  (void)in_sizes; (void)n_in; (void)out_size; (void)ws_size;
  const float* x    = (const float*)d_in[0];
  const float* gw   = (const float*)d_in[1];
  const int*   gup  = (const int*)d_in[2];
  const float* gus  = (const float*)d_in[3];
  const int*   dp   = (const int*)d_in[4];
  const float* dsc  = (const float*)d_in[5];
  const int*   sgup = (const int*)d_in[6];
  const float* sgus = (const float*)d_in[7];
  const int*   sdp  = (const int*)d_in[8];
  const float* sdsc = (const float*)d_in[9];
  float* out = (float*)d_out;

  char* ws = (char*)d_ws;
  int*   counts      = (int*)(ws + 0);
  int*   offsets     = (int*)(ws + 64);
  int*   tok_idx     = (int*)(ws + 128);
  float* tok_w       = (float*)(ws + 128 + 16384);
  int*   tokens_list = (int*)(ws + 128 + 2 * 16384);
  float* weight_list = (float*)(ws + 128 + 3 * 16384);
  float* act_shared  = (float*)(ws + 65792);                       // [NT, ID]
  float* act_routed  = act_shared + (size_t)NT * ID;               // [NT*TOPK, ID]

  route_kernel<<<dim3(NT / 256), 256, 0, stream>>>(x, gw, tok_idx, tok_w);
  compact_kernel<<<dim3(1), 1024, 0, stream>>>(tok_idx, tok_w, counts, offsets,
                                               tokens_list, weight_list);
  gateup_kernel<true><<<dim3(ID / 128, NT / 32, 1), 256, 0, stream>>>(
      x, sgup, sgus, nullptr, nullptr, nullptr, act_shared);
  gateup_kernel<false><<<dim3(ID / 128, NT / 32, TE), 256, 0, stream>>>(
      x, gup, gus, counts, offsets, tokens_list, act_routed);
  // shared down WRITES out first, routed down atomically accumulates after
  down_kernel<true><<<dim3(HD / 128, NT / 32, 1), 256, 0, stream>>>(
      act_shared, sdp, sdsc, nullptr, nullptr, nullptr, nullptr, out);
  down_kernel<false><<<dim3(HD / 128, NT / 32, TE), 256, 0, stream>>>(
      act_routed, dp, dsc, counts, offsets, tokens_list, weight_list, out);
}
